// RobertaEncoder_52278341927450
// MI455X (gfx1250) — compile-verified, matrix operand order checked by reading, not run on hardware
//
#include <hip/hip_runtime.h>

// RoBERTa encoder (L=2, B=2, S=2048, H=1024, NH=16, DH=64, FF=4096) on gfx1250.
// bf16 operands converted ONCE; all matmuls via v_wmma_f32_16x16x32_bf16;
// GEMM tiles staged to LDS with global_load_async_to_lds_b128 when available.

typedef __attribute__((ext_vector_type(16))) __bf16 v16bf;
typedef __attribute__((ext_vector_type(8)))  __bf16 v8bf;
typedef __attribute__((ext_vector_type(8)))  float  v8f;
typedef __attribute__((ext_vector_type(4)))  int    v4i;

#define WMMA_BF16(a, b, c) \
  __builtin_amdgcn_wmma_f32_16x16x32_bf16(false, (a), false, (b), (short)0, (c), false, false)

// float -> bf16 RNE via integer ops (used only in one-time conversions/epilogues).
__device__ __forceinline__ __bf16 f2bf(float f) {
  unsigned u = __builtin_bit_cast(unsigned, f);
  unsigned r = (u + 0x7FFFu + ((u >> 16) & 1u)) >> 16;
  unsigned short h = (unsigned short)r;
  return __builtin_bit_cast(__bf16, h);
}

// Load a 16x32-layout A/B fragment half-pair: lane owns one row; takes row base
// pointer and kb8 = 8*(lane>=16). Two aligned 16B vector loads.
__device__ __forceinline__ v16bf ld_frag(const __bf16* __restrict__ row, int kb8) {
  v8bf lo = *(const v8bf*)(row + kb8);
  v8bf hi = *(const v8bf*)(row + 16 + kb8);
  return __builtin_shufflevector(lo, hi, 0, 1, 2, 3, 4, 5, 6, 7, 8, 9, 10, 11, 12, 13, 14, 15);
}

// ---- async global->LDS 16B copy (CDNA5 path), plain copy fallback ----------
#if defined(__has_builtin)
#if __has_builtin(__builtin_amdgcn_global_load_async_to_lds_b128)
#define HAVE_ASYNC_LDS 1
#endif
#endif
#ifndef HAVE_ASYNC_LDS
#define HAVE_ASYNC_LDS 0
#endif

typedef __attribute__((address_space(1))) v4i gv4i;
typedef __attribute__((address_space(3))) v4i lv4i;

__device__ __forceinline__ void cp16_g2l(const __bf16* __restrict__ g, __bf16* __restrict__ l) {
#if HAVE_ASYNC_LDS
  gv4i* gp = (gv4i*)(unsigned long long)(uintptr_t)g;
  lv4i* lp = (lv4i*)(unsigned)(uintptr_t)l;
  __builtin_amdgcn_global_load_async_to_lds_b128(gp, lp, 0, 0);
#else
  *(uint4*)l = *(const uint4*)g;
#endif
}

__device__ __forceinline__ void async_tile_wait() {
#if HAVE_ASYNC_LDS
  __asm__ volatile("s_wait_asynccnt 0" ::: "memory");
#endif
}

// ---------------------------------------------------------------------------
// one-time fp32 -> bf16 conversion
// ---------------------------------------------------------------------------
__global__ __launch_bounds__(256) void cvt_bf16_kernel(const float* __restrict__ src,
                                                       __bf16* __restrict__ dst, int n) {
  const int i = (blockIdx.x * 256 + threadIdx.x) * 8;
  if (i + 8 <= n) {
#pragma unroll
    for (int j = 0; j < 8; ++j) dst[i + j] = f2bf(src[i + j]);
  }
}

// one-time W[K,N] fp32 -> Wt[N,K] bf16 (transpose via LDS tile)
__global__ __launch_bounds__(256) void transpose_cvt_kernel(const float* __restrict__ W,
                                                            __bf16* __restrict__ Wt,
                                                            int K, int N) {
  __shared__ __bf16 t[32][33];
  const int k0 = blockIdx.y * 32, n0 = blockIdx.x * 32;
  const int c = threadIdx.x & 31, rq = threadIdx.x >> 5;
#pragma unroll
  for (int i = 0; i < 4; ++i) {
    const int r = rq * 4 + i;
    t[c][r] = f2bf(W[(size_t)(k0 + r) * N + n0 + c]);
  }
  __syncthreads();
#pragma unroll
  for (int i = 0; i < 4; ++i) {
    const int r = rq * 4 + i;
    Wt[(size_t)(n0 + r) * K + k0 + c] = t[r][c];
  }
}

// ---------------------------------------------------------------------------
// GEMM: out = act(A[M,K](bf16) @ WT[N,K](bf16)^T + bias) (+ res fp32)
// Block 128 threads (4 waves), tile 128x64, BK=32; wave = 32 rows x 64 cols.
// OUTMODE: 0 = fp32, 1 = bf16, 2 = bf16 scattered to V^T [B,NH,DH,S]
// ---------------------------------------------------------------------------
template <int OUTMODE, bool GELU, bool RES>
__global__ __launch_bounds__(128) void gemm_bf16_kernel(
    const __bf16* __restrict__ A, const __bf16* __restrict__ WT,
    const float* __restrict__ bias, const float* __restrict__ res,
    void* __restrict__ outv, int M, int N, int K) {
  __shared__ __bf16 As[128][32];
  __shared__ __bf16 Bs[64][32];
  const int tid = threadIdx.x;
  const int wave = tid >> 5, lane = tid & 31;
  const int lr = lane & 15, hi = lane >> 4, kb8 = hi * 8;
  const int m0 = blockIdx.y * 128, n0 = blockIdx.x * 64;

  v8f acc[2][4] = {};

  for (int kk = 0; kk < K; kk += 32) {
    // A tile: 128x32 bf16 = 512 x 16B chunks; B tile: 64x32 = 256 chunks
#pragma unroll
    for (int i = 0; i < 4; ++i) {
      const int ch = tid + 128 * i;
      const int row = ch >> 2, ks = (ch & 3) * 8;
      cp16_g2l(A + (size_t)(m0 + row) * K + kk + ks, &As[row][ks]);
    }
#pragma unroll
    for (int i = 0; i < 2; ++i) {
      const int ch = tid + 128 * i;
      const int row = ch >> 2, ks = (ch & 3) * 8;
      cp16_g2l(WT + (size_t)(n0 + row) * K + kk + ks, &Bs[row][ks]);
    }
    async_tile_wait();
    __syncthreads();

    const v16bf af0 = ld_frag(&As[wave * 32 + lr][0], kb8);
    const v16bf af1 = ld_frag(&As[wave * 32 + 16 + lr][0], kb8);
#pragma unroll
    for (int nt = 0; nt < 4; ++nt) {
      const v16bf bf = ld_frag(&Bs[nt * 16 + lr][0], kb8);
      acc[0][nt] = WMMA_BF16(af0, bf, acc[0][nt]);
      acc[1][nt] = WMMA_BF16(af1, bf, acc[1][nt]);
    }
    __syncthreads();
  }

#pragma unroll
  for (int rg = 0; rg < 2; ++rg) {
#pragma unroll
    for (int nt = 0; nt < 4; ++nt) {
      const int col = n0 + nt * 16 + lr;
      const float bv = bias[col];
#pragma unroll
      for (int j = 0; j < 8; ++j) {
        const int row = m0 + wave * 32 + rg * 16 + hi * 8 + j;
        float v = acc[rg][nt][j] + bv;
        if (GELU) v = 0.5f * v * (1.0f + erff(v * 0.70710678118654752f));
        if (RES) v += res[(size_t)row * N + col];
        if (OUTMODE == 0) {
          ((float*)outv)[(size_t)row * N + col] = v;
        } else if (OUTMODE == 1) {
          ((__bf16*)outv)[(size_t)row * N + col] = f2bf(v);
        } else {
          constexpr int S = 2048, NH = 16;
          const int b = row >> 11, s = row & (S - 1);
          const int head = col >> 6, dh = col & 63;
          ((__bf16*)outv)[(((size_t)b * NH + head) * 64 + dh) * S + s] = f2bf(v);
        }
      }
    }
  }
}

// ---------------------------------------------------------------------------
// Flash attention: wave = 16 query rows of one (batch, head), key blocks of 32.
// Q,K bf16 [B,S,H]; V bf16 pre-transposed [B,NH,DH,S]; ctx out bf16 [B,S,H].
// ---------------------------------------------------------------------------
__global__ __launch_bounds__(128) void attention_kernel(
    const __bf16* __restrict__ qb, const __bf16* __restrict__ kb,
    const __bf16* __restrict__ vt, const float* __restrict__ mask,
    __bf16* __restrict__ ctx) {
  constexpr int S = 2048, H = 1024, NH = 16;
  __shared__ __bf16 Plds[4][16 * 32];
  const int tid = threadIdx.x, wave = tid >> 5, lane = tid & 31;
  const int lr = lane & 15, hi = lane >> 4, kb8 = hi * 8;
  const int head = blockIdx.y, batch = blockIdx.z;
  const int q0 = blockIdx.x * 64 + wave * 16;
  const size_t bo = (size_t)batch * S * H + (size_t)head * 64;
  const __bf16* Q = qb + bo;
  const __bf16* Kp = kb + bo;
  const __bf16* Vp = vt + ((size_t)batch * NH + head) * (size_t)64 * S;  // [dh][s]
  const float* mk = mask + (size_t)batch * S;

  const v16bf qa0 = ld_frag(Q + (size_t)(q0 + lr) * H, kb8);
  const v16bf qa1 = ld_frag(Q + (size_t)(q0 + lr) * H + 32, kb8);

  float m_i[8], l_i[8];
  v8f o[4] = {};
#pragma unroll
  for (int j = 0; j < 8; ++j) { m_i[j] = -3.0e38f; l_i[j] = 0.f; }
  __bf16* pw = &Plds[wave][0];

  for (int kp0 = 0; kp0 < S; kp0 += 32) {
    v8f sacc[2] = {};
#pragma unroll
    for (int g = 0; g < 2; ++g) {
      const __bf16* kr = Kp + (size_t)(kp0 + g * 16 + lr) * H;
      sacc[g] = WMMA_BF16(qa0, ld_frag(kr, kb8), sacc[g]);
      sacc[g] = WMMA_BF16(qa1, ld_frag(kr + 32, kb8), sacc[g]);
    }

    const float mk0 = mk[kp0 + lr], mk1 = mk[kp0 + 16 + lr];
    float ps0[8], ps1[8];
#pragma unroll
    for (int j = 0; j < 8; ++j) {
      ps0[j] = sacc[0][j] * 0.125f + mk0;
      ps1[j] = sacc[1][j] * 0.125f + mk1;
      float mx = fmaxf(ps0[j], ps1[j]);
#pragma unroll
      for (int d = 1; d < 16; d <<= 1) mx = fmaxf(mx, __shfl_xor(mx, d, 32));
      const float nm = fmaxf(m_i[j], mx);
      ps0[j] = __expf(ps0[j] - nm);
      ps1[j] = __expf(ps1[j] - nm);
      float sm = ps0[j] + ps1[j];
#pragma unroll
      for (int d = 1; d < 16; d <<= 1) sm += __shfl_xor(sm, d, 32);
      const float al = __expf(m_i[j] - nm);
      l_i[j] = l_i[j] * al + sm;
      m_i[j] = nm;
#pragma unroll
      for (int dt = 0; dt < 4; ++dt) o[dt][j] *= al;
    }

    // stage P (accumulator layout -> row-major 16x32 bf16), reload as A-frag
#pragma unroll
    for (int j = 0; j < 8; ++j) {
      pw[(hi * 8 + j) * 32 + lr] = f2bf(ps0[j]);
      pw[(hi * 8 + j) * 32 + 16 + lr] = f2bf(ps1[j]);
    }
    __asm__ volatile("s_wait_dscnt 0" ::: "memory");
    const v16bf pa = ld_frag(&pw[lr * 32], kb8);

    // P[16x32] @ V[32x64] ; V^T rows are contiguous in s
#pragma unroll
    for (int dt = 0; dt < 4; ++dt) {
      const v16bf vf = ld_frag(Vp + (size_t)(dt * 16 + lr) * S + kp0, kb8);
      o[dt] = WMMA_BF16(pa, vf, o[dt]);
    }
  }

#pragma unroll
  for (int dt = 0; dt < 4; ++dt) {
    const int col = dt * 16 + lr;
#pragma unroll
    for (int j = 0; j < 8; ++j) {
      const int row = q0 + hi * 8 + j;
      ctx[bo + (size_t)row * H + col] = f2bf(o[dt][j] / l_i[j]);
    }
  }
}

// ---------------------------------------------------------------------------
// LayerNorm over H=1024: one wave per row; emits fp32 and bf16 copies.
// ---------------------------------------------------------------------------
__global__ __launch_bounds__(256) void layernorm_kernel(
    const float* __restrict__ in, const float* __restrict__ g,
    const float* __restrict__ b, float* __restrict__ outf,
    __bf16* __restrict__ outb) {
  const int wave = threadIdx.x >> 5, lane = threadIdx.x & 31;
  const size_t row = (size_t)blockIdx.x * 8 + wave;
  const float* x = in + row * 1024;
  float v[32];
  float s = 0.f;
#pragma unroll
  for (int i = 0; i < 32; ++i) { v[i] = x[lane + i * 32]; s += v[i]; }
#pragma unroll
  for (int d = 1; d < 32; d <<= 1) s += __shfl_xor(s, d, 32);
  const float mean = s * (1.0f / 1024.0f);
  float vs = 0.f;
#pragma unroll
  for (int i = 0; i < 32; ++i) { const float df = v[i] - mean; vs += df * df; }
#pragma unroll
  for (int d = 1; d < 32; d <<= 1) vs += __shfl_xor(vs, d, 32);
  const float rstd = rsqrtf(vs * (1.0f / 1024.0f) + 1e-5f);
#pragma unroll
  for (int i = 0; i < 32; ++i) {
    const int c = lane + i * 32;
    const float y = (v[i] - mean) * rstd * g[c] + b[c];
    outf[row * 1024 + c] = y;
    outb[row * 1024 + c] = f2bf(y);
  }
}

// ---------------------------------------------------------------------------
extern "C" void kernel_launch(void* const* d_in, const int* in_sizes, int n_in,
                              void* d_out, int out_size, void* d_ws, size_t ws_size,
                              hipStream_t stream) {
  (void)in_sizes; (void)n_in; (void)out_size; (void)ws_size;
  constexpr int L = 2, B = 2, S = 2048, H = 1024, FF = 4096;
  constexpr int M = B * S;
  constexpr size_t MH = (size_t)M * H;
  constexpr size_t MF = (size_t)M * FF;

  const float* x0   = (const float*)d_in[0];
  const float* mask = (const float*)d_in[1];
  const float* Wq   = (const float*)d_in[2];
  const float* bq   = (const float*)d_in[3];
  const float* Wk   = (const float*)d_in[4];
  const float* bk   = (const float*)d_in[5];
  const float* Wv   = (const float*)d_in[6];
  const float* bv   = (const float*)d_in[7];
  const float* Wao  = (const float*)d_in[8];
  const float* bao  = (const float*)d_in[9];
  const float* g1   = (const float*)d_in[10];
  const float* b1   = (const float*)d_in[11];
  const float* Wi   = (const float*)d_in[12];
  const float* bi   = (const float*)d_in[13];
  const float* Wo   = (const float*)d_in[14];
  const float* bo   = (const float*)d_in[15];
  const float* g2   = (const float*)d_in[16];
  const float* b2   = (const float*)d_in[17];

  // --- workspace bump allocator (256B aligned regions) ---
  char* p = (char*)d_ws;
  auto alloc = [&](size_t bytes) -> char* {
    char* r = p;
    p += (bytes + 255) & ~(size_t)255;
    return r;
  };
  float*  yB  = (float*)alloc(MH * 4);
  float*  x1f = (float*)alloc(MH * 4);
  float*  xf  = (float*)alloc(MH * 4);
  __bf16* xb   = (__bf16*)alloc(MH * 2);
  __bf16* qB   = (__bf16*)alloc(MH * 2);
  __bf16* kB   = (__bf16*)alloc(MH * 2);
  __bf16* vT   = (__bf16*)alloc(MH * 2);
  __bf16* ctxb = (__bf16*)alloc(MH * 2);
  __bf16* x1b  = (__bf16*)alloc(MH * 2);
  __bf16* h1b  = (__bf16*)alloc(MF * 2);
  __bf16* WqT[L]; __bf16* WkT[L]; __bf16* WvT[L]; __bf16* WaoT[L];
  __bf16* WiT[L]; __bf16* WoT[L];
  for (int l = 0; l < L; ++l) {
    WqT[l]  = (__bf16*)alloc((size_t)H * H * 2);
    WkT[l]  = (__bf16*)alloc((size_t)H * H * 2);
    WvT[l]  = (__bf16*)alloc((size_t)H * H * 2);
    WaoT[l] = (__bf16*)alloc((size_t)H * H * 2);
    WiT[l]  = (__bf16*)alloc((size_t)H * FF * 2);
    WoT[l]  = (__bf16*)alloc((size_t)FF * H * 2);
  }

  const dim3 blk(128), blk256(256);
  const dim3 gHH(H / 64, M / 128);
  const dim3 gHF(FF / 64, M / 128);
  const dim3 gAttn(S / 64, 16, B);
  const dim3 gLN(M / 8);
  const dim3 gtHH(H / 32, H / 32);
  const dim3 gtHF(FF / 32, H / 32);
  const dim3 gtFH(H / 32, FF / 32);

  // one-time conversions
  cvt_bf16_kernel<<<dim3(MH / 2048), blk256, 0, stream>>>(x0, xb, (int)MH);
  for (int l = 0; l < L; ++l) {
    const size_t oHH = (size_t)l * H * H, oHF = (size_t)l * H * FF;
    transpose_cvt_kernel<<<gtHH, blk256, 0, stream>>>(Wq + oHH, WqT[l], H, H);
    transpose_cvt_kernel<<<gtHH, blk256, 0, stream>>>(Wk + oHH, WkT[l], H, H);
    transpose_cvt_kernel<<<gtHH, blk256, 0, stream>>>(Wv + oHH, WvT[l], H, H);
    transpose_cvt_kernel<<<gtHH, blk256, 0, stream>>>(Wao + oHH, WaoT[l], H, H);
    transpose_cvt_kernel<<<gtHF, blk256, 0, stream>>>(Wi + oHF, WiT[l], H, FF);
    transpose_cvt_kernel<<<gtFH, blk256, 0, stream>>>(Wo + oHF, WoT[l], FF, H);
  }

  for (int l = 0; l < L; ++l) {
    const float* Xf = (l == 0) ? x0 : xf;
    const size_t oH = (size_t)l * H, oF = (size_t)l * FF;

    gemm_bf16_kernel<1, false, false><<<gHH, blk, 0, stream>>>(xb, WqT[l], bq + oH, nullptr, qB, M, H, H);
    gemm_bf16_kernel<1, false, false><<<gHH, blk, 0, stream>>>(xb, WkT[l], bk + oH, nullptr, kB, M, H, H);
    gemm_bf16_kernel<2, false, false><<<gHH, blk, 0, stream>>>(xb, WvT[l], bv + oH, nullptr, vT, M, H, H);
    attention_kernel<<<gAttn, blk, 0, stream>>>(qB, kB, vT, mask, ctxb);
    gemm_bf16_kernel<0, false, true><<<gHH, blk, 0, stream>>>(ctxb, WaoT[l], bao + oH, Xf, yB, M, H, H);
    layernorm_kernel<<<gLN, blk256, 0, stream>>>(yB, g1 + oH, b1 + oH, x1f, x1b);
    gemm_bf16_kernel<1, true, false><<<gHF, blk, 0, stream>>>(x1b, WiT[l], bi + oF, nullptr, h1b, M, FF, H);
    gemm_bf16_kernel<0, false, true><<<gHH, blk, 0, stream>>>(h1b, WoT[l], bo + oH, x1f, yB, M, H, FF);
    float* xoutf = (l == L - 1) ? (float*)d_out : xf;
    layernorm_kernel<<<gLN, blk256, 0, stream>>>(yB, g2 + oH, b2 + oH, xoutf, xb);
  }
}
